// LSTMPolicyNetwork_5428838662581
// MI455X (gfx1250) — matrix operand order repaired and checked by reading, not compiled, not run on hardware
//
#include <hip/hip_runtime.h>
#include <hip/hip_bf16.h>

typedef __attribute__((ext_vector_type(16))) _Float16 v16h;
typedef __attribute__((ext_vector_type(8)))  _Float16 v8h;
typedef __attribute__((ext_vector_type(8)))  float    v8f;

#define HID 16
#define ACT 4
#define NB  524288

// ---------------------------------------------------------------------------
// WMMA wrapper: D = A(16x32 f16) * B(32x16 f16) + C(16x16 f32)
// ---------------------------------------------------------------------------
__device__ __forceinline__ v8f wmma16(v16h a, v16h b, v8f c) {
  return __builtin_amdgcn_wmma_f32_16x16x32_f16(false, a, false, b,
                                                (short)0, c, false, false);
}

// Load one pre-packed B fragment (32 bytes per lane, L2-broadcast friendly)
__device__ __forceinline__ v16h load_bfrag(const _Float16* packed, int f, int lane) {
  return *(const v16h*)(packed + ((size_t)(f * 32 + lane)) * 16);
}

// A-fragment load, full K=32 chunk starting at kbase in a (16 x stride) f16 tile
__device__ __forceinline__ v16h a_load32(const _Float16* st, int stride, int kbase, int lane) {
  const int m = lane & 15, hi = lane >> 4;
  v8h lo = *(const v8h*)(st + m * stride + kbase + hi * 8);
  v8h hh = *(const v8h*)(st + m * stride + kbase + 16 + hi * 8);
  v16h a;
#pragma unroll
  for (int i = 0; i < 8; ++i) { a[i] = lo[i]; a[8 + i] = hh[i]; }
  return a;
}

// Raw hardware reciprocal: single v_rcp_f32 (~1 ulp), no IEEE div expansion
__device__ __forceinline__ float rcp_(float x) { return __builtin_amdgcn_rcpf(x); }

__device__ __forceinline__ float sigmoidf_(float x) {
  return rcp_(1.0f + __expf(-x));
}
__device__ __forceinline__ float tanhf_(float x) {
  float t = __expf(2.0f * x);
  return (t - 1.0f) * rcp_(t + 1.0f);
}

// ---------------------------------------------------------------------------
// Pack kernel: weights -> WMMA B-fragment layout (f16).
// B-matrix (32x16 f16) layout: lane l -> N = l&15; element i -> K = 8*(l>>4) + i (i<8), 16 + 8*(l>>4) + (i-8) (i>=8)
// Fragments:
//   0..1   : combined gate B for r,z:  K<16 -> Wih_j ; K>=16 -> Whh_j
//   2      : Wih (n-gate)  (K<16 valid)
//   3      : Whh (n-gate)  (K<16 valid)
//   4..11  : Wv1.T  8 N-tiles (K<16 valid)
//   12..27 : Wv2.T  [t=0..3][c=0..3]
//   28..31 : Wv3.T  [t=0..1][c=0..1]
//   32..33 : Wv4.T  [t=0..1]
// ---------------------------------------------------------------------------
__global__ void pack_weights(const float* Wih, const float* Whh,
                             const float* Wv1, const float* Wv2,
                             const float* Wv3, const float* Wv4, _Float16* packed) {
  const int lane = threadIdx.x & 31;
  const int n = lane & 15, hi = lane >> 4;
  auto put = [&](int f, int i, float v) {
    packed[((size_t)(f * 32 + lane)) * 16 + i] = (_Float16)v;
  };
  for (int i = 0; i < 16; ++i) {
    const int k = (i < 8) ? (hi * 8 + i) : (16 + hi * 8 + (i - 8));
    for (int j = 0; j < 2; ++j)  // combined r,z gates
      put(j, i, (k < 16) ? Wih[(j * 16 + n) * HID + k]
                         : Whh[(j * 16 + n) * HID + (k - 16)]);
    put(2, i, (k < 16) ? Wih[(32 + n) * HID + k] : 0.0f);
    put(3, i, (k < 16) ? Whh[(32 + n) * HID + k] : 0.0f);
    for (int t = 0; t < 8; ++t)
      put(4 + t, i, (k < 16) ? Wv1[(t * 16 + n) * HID + k] : 0.0f);
    for (int t = 0; t < 4; ++t)
      for (int c = 0; c < 4; ++c)
        put(12 + t * 4 + c, i, Wv2[(t * 16 + n) * 128 + c * 32 + k]);
    for (int t = 0; t < 2; ++t)
      for (int c = 0; c < 2; ++c)
        put(28 + t * 2 + c, i, Wv3[(t * 16 + n) * 64 + c * 32 + k]);
    for (int t = 0; t < 2; ++t)
      put(32 + t, i, Wv4[(t * 16 + n) * 32 + k]);
  }
}

// ---------------------------------------------------------------------------
// Decoder kernel: 16 GRU steps per 16-row tile, one tile per wave.
// Per step: 4 WMMAs + one h C/D->A LDS transpose. e and logits are computed
// per-lane (K=4 sides), with a cross-group shfl_xor(16) logits reduction.
// ---------------------------------------------------------------------------
__global__ void __launch_bounds__(256) decoder_kernel(
    const float* __restrict__ features, const float* __restrict__ We,
    const float* __restrict__ be, const float* __restrict__ bih,
    const float* __restrict__ bhh, const float* __restrict__ Wout,
    const float* __restrict__ bout, const _Float16* __restrict__ packed,
    float* __restrict__ actor_out) {
  __shared__ __align__(16) _Float16 stage[8][16 * 16];  // per-wave h tile

  const int wave = threadIdx.x >> 5;
  const int lane = threadIdx.x & 31;
  const int n = lane & 15, hi = lane >> 4, m = lane & 15;
  const int base = blockIdx.x * 128 + wave * 16;
  _Float16* st = stage[wave];

  // Resident gate B fragments
  const v16h B_G0  = load_bfrag(packed, 0, lane);  // [Wih_r ; Whh_r]
  const v16h B_G1  = load_bfrag(packed, 1, lane);  // [Wih_z ; Whh_z]
  const v16h B_Wi2 = load_bfrag(packed, 2, lane);  // Wih_n
  const v16h B_Wh2 = load_bfrag(packed, 3, lane);  // Whh_n

  // Per-lane (column n) gate biases; r,z biases pre-summed
  const float bg0 = bih[n] + bhh[n];
  const float bg1 = bih[16 + n] + bhh[16 + n];
  const float bi2 = bih[32 + n];
  const float bh2 = bhh[32 + n];

  // Lane-group-resident encoder slice: e[k] = relu(be[k] + sum_j x[j]*We[k][j]),
  // this lane covers k = hi*8 + i
  float WeL[8][4], beL[8];
#pragma unroll
  for (int i = 0; i < 8; ++i) {
    const int k = hi * 8 + i;
    beL[i] = be[k];
#pragma unroll
    for (int j = 0; j < ACT; ++j) WeL[i][j] = We[k * ACT + j];
  }
  // Lane-group-resident output head slice: logits[j] = sum_k h[k]*Wout[j][k]
  float WoL[4][8], boutv[4];
#pragma unroll
  for (int j = 0; j < ACT; ++j) {
    boutv[j] = bout[j];
#pragma unroll
    for (int i = 0; i < 8; ++i) WoL[j][i] = Wout[j * HID + hi * 8 + i];
  }

  // h init = features: C/D-layout values + A-layout halves (lane m, k=hi*8+i)
  v8f h_val;
#pragma unroll
  for (int r = 0; r < 8; ++r)
    h_val[r] = features[(size_t)(base + r + hi * 8) * HID + n];
  v8h hh8;
  {
    const float* fp = features + (size_t)(base + m) * HID + hi * 8;
#pragma unroll
    for (int i = 0; i < 8; ++i) hh8[i] = (_Float16)fp[i];
  }

  float x0 = 0.0f, x1 = 0.0f, x2 = 0.0f, x3 = 0.0f;  // x starts at zeros
  int target = 0;
  const v8f zero = {};

#pragma unroll 1
  for (int step = 0; step < 16; ++step) {
    // e = relu(x @ We.T + be), this lane's 8 of 16 outputs, straight to f16
    v8h e8;
#pragma unroll
    for (int i = 0; i < 8; ++i) {
      float v = beL[i] + x0 * WeL[i][0] + x1 * WeL[i][1] +
                x2 * WeL[i][2] + x3 * WeL[i][3];
      e8[i] = (_Float16)(v > 0.0f ? v : 0.0f);
    }

    // A fragments: combined [e|h] (K=32), e-only, h-only (K=16)
    v16h comb, efrag = {}, hfrag = {};
#pragma unroll
    for (int i = 0; i < 8; ++i) {
      comb[i] = e8[i]; comb[8 + i] = hh8[i];
      efrag[i] = e8[i]; hfrag[i] = hh8[i];
    }

    v8f g0  = wmma16(comb,  B_G0,  zero);  // ir + hr
    v8f g1  = wmma16(comb,  B_G1,  zero);  // iz + hz
    v8f gi2 = wmma16(efrag, B_Wi2, zero);  // inn
    v8f gh2 = wmma16(hfrag, B_Wh2, zero);  // hn

#pragma unroll
    for (int r = 0; r < 8; ++r) {
      float rr = sigmoidf_(g0[r] + bg0);
      float zz = sigmoidf_(g1[r] + bg1);
      float nn = tanhf_((gi2[r] + bi2) + rr * (gh2[r] + bh2));
      h_val[r] = (1.0f - zz) * nn + zz * h_val[r];
    }

    // h2: C/D layout -> LDS f16 tile -> reload this lane's A-layout halves
#pragma unroll
    for (int r = 0; r < 8; ++r)
      st[(r + hi * 8) * 16 + n] = (_Float16)h_val[r];
    hh8 = *(const v8h*)(st + m * 16 + hi * 8);

    // logits = h2 @ Wout.T + bout : per-lane partial over 8 K, pair-reduce
    float hf[8];
#pragma unroll
    for (int i = 0; i < 8; ++i) hf[i] = (float)hh8[i];
    float lg[4];
#pragma unroll
    for (int j = 0; j < ACT; ++j) {
      float s = hf[0] * WoL[j][0] + hf[1] * WoL[j][1] + hf[2] * WoL[j][2] +
                hf[3] * WoL[j][3] + hf[4] * WoL[j][4] + hf[5] * WoL[j][5] +
                hf[6] * WoL[j][6] + hf[7] * WoL[j][7];
      s += __shfl_xor(s, 16, 32);  // combine K halves across lane groups
      lg[j] = s + boutv[j];
    }
    x0 = lg[0]; x1 = lg[1]; x2 = lg[2]; x3 = lg[3];

    // per-lane softmax over the 4 logits
    float mx = fmaxf(fmaxf(x0, x1), fmaxf(x2, x3));
    float e0 = __expf(x0 - mx), e1 = __expf(x1 - mx);
    float e2 = __expf(x2 - mx), e3 = __expf(x3 - mx);
    float inv = rcp_(e0 + e1 + e2 + e3);
    float p0 = e0 * inv, p1 = e1 * inv, p2 = e2 * inv, p3 = e3 * inv;

    if (step == 1)  // obs_count -> target
      target = (int)(rintf(p0 * 2.0f) + rintf(p1 * 5.0f) + rintf(p2 * 7.0f));

    float w = 1.0f;
    if (step >= 2) w = ((step - 2) < target) ? 1.0f : 0.0f;

    if (lane < 16) {
      float4 o = make_float4(p0 * w, p1 * w, p2 * w, p3 * w);
      *(float4*)(actor_out + (size_t)(base + m) * 64 + step * 4) = o;
    }
  }
}

// ---------------------------------------------------------------------------
// Value MLP kernel: 16 -> 128 -> 64 -> 32 -> 32, one 16-row tile per wave.
// ---------------------------------------------------------------------------
__global__ void __launch_bounds__(256) value_kernel(
    const float* __restrict__ features, const float* __restrict__ bv1,
    const float* __restrict__ bv2, const float* __restrict__ bv3,
    const float* __restrict__ bv4, const _Float16* __restrict__ packed,
    float* __restrict__ v_out) {
  __shared__ __align__(16) _Float16 vstage[8][16 * 128];  // per-wave activations

  const int wave = threadIdx.x >> 5;
  const int lane = threadIdx.x & 31;
  const int n = lane & 15, hi = lane >> 4, m = lane & 15;
  const int base = blockIdx.x * 128 + wave * 16;
  _Float16* st = vstage[wave];
  const v8f zero = {};

  // features as A fragment (K=16 valid)
  v16h a0 = {};
  {
    const float* fp = features + (size_t)(base + m) * HID + hi * 8;
#pragma unroll
    for (int i = 0; i < 8; ++i) a0[i] = (_Float16)fp[i];
  }

  // layer 1: 16 -> 128
#pragma unroll
  for (int t = 0; t < 8; ++t) {
    v8f acc = wmma16(a0, load_bfrag(packed, 4 + t, lane), zero);
    float b = bv1[t * 16 + n];
#pragma unroll
    for (int r = 0; r < 8; ++r) {
      float v = acc[r] + b; v = v > 0.0f ? v : 0.0f;
      st[(r + hi * 8) * 128 + t * 16 + n] = (_Float16)v;
    }
  }
  v16h a1[4];
#pragma unroll
  for (int c = 0; c < 4; ++c) a1[c] = a_load32(st, 128, c * 32, lane);

  // layer 2: 128 -> 64
#pragma unroll
  for (int t = 0; t < 4; ++t) {
    v8f acc = zero;
#pragma unroll
    for (int c = 0; c < 4; ++c)
      acc = wmma16(a1[c], load_bfrag(packed, 12 + t * 4 + c, lane), acc);
    float b = bv2[t * 16 + n];
#pragma unroll
    for (int r = 0; r < 8; ++r) {
      float v = acc[r] + b; v = v > 0.0f ? v : 0.0f;
      st[(r + hi * 8) * 64 + t * 16 + n] = (_Float16)v;
    }
  }
  v16h a2[2];
#pragma unroll
  for (int c = 0; c < 2; ++c) a2[c] = a_load32(st, 64, c * 32, lane);

  // layer 3: 64 -> 32
#pragma unroll
  for (int t = 0; t < 2; ++t) {
    v8f acc = zero;
#pragma unroll
    for (int c = 0; c < 2; ++c)
      acc = wmma16(a2[c], load_bfrag(packed, 28 + t * 2 + c, lane), acc);
    float b = bv3[t * 16 + n];
#pragma unroll
    for (int r = 0; r < 8; ++r) {
      float v = acc[r] + b; v = v > 0.0f ? v : 0.0f;
      st[(r + hi * 8) * 32 + t * 16 + n] = (_Float16)v;
    }
  }
  v16h a3 = a_load32(st, 32, 0, lane);

  // layer 4: 32 -> 32, write out
#pragma unroll
  for (int t = 0; t < 2; ++t) {
    v8f acc = wmma16(a3, load_bfrag(packed, 32 + t, lane), zero);
    float b = bv4[t * 16 + n];
#pragma unroll
    for (int r = 0; r < 8; ++r) {
      float v = acc[r] + b; v = v > 0.0f ? v : 0.0f;
      v_out[(size_t)(base + r + hi * 8) * 32 + t * 16 + n] = v;
    }
  }
}

// ---------------------------------------------------------------------------
extern "C" void kernel_launch(void* const* d_in, const int* in_sizes, int n_in,
                              void* d_out, int out_size, void* d_ws, size_t ws_size,
                              hipStream_t stream) {
  const float* features = (const float*)d_in[0];
  const float* We   = (const float*)d_in[1];
  const float* be   = (const float*)d_in[2];
  const float* Wih  = (const float*)d_in[3];
  const float* Whh  = (const float*)d_in[4];
  const float* bih  = (const float*)d_in[5];
  const float* bhh  = (const float*)d_in[6];
  const float* Wout = (const float*)d_in[7];
  const float* bout = (const float*)d_in[8];
  const float* Wv1  = (const float*)d_in[9];
  const float* bv1  = (const float*)d_in[10];
  const float* Wv2  = (const float*)d_in[11];
  const float* bv2  = (const float*)d_in[12];
  const float* Wv3  = (const float*)d_in[13];
  const float* bv3  = (const float*)d_in[14];
  const float* Wv4  = (const float*)d_in[15];
  const float* bv4  = (const float*)d_in[16];

  _Float16* packed = (_Float16*)d_ws;
  float* actor_out = (float*)d_out;                    // B * 16 * 1 * 4
  float* v_out     = (float*)d_out + (size_t)NB * 64;  // B * 32

  pack_weights<<<1, 32, 0, stream>>>(Wih, Whh, Wv1, Wv2, Wv3, Wv4, packed);

  const int grid = NB / 128;  // 128 rows per block (8 waves x 16-row tiles)
  decoder_kernel<<<grid, 256, 0, stream>>>(features, We, be, bih, bhh, Wout,
                                           bout, packed, actor_out);
  value_kernel<<<grid, 256, 0, stream>>>(features, bv1, bv2, bv3, bv4, packed, v_out);
}
